// MEGANCore_9088150798343
// MI455X (gfx1250) — matrix operand order repaired and checked by reading, not compiled, not run on hardware
//
#include <hip/hip_runtime.h>
#include <hip/hip_bf16.h>

typedef __attribute__((ext_vector_type(16))) __bf16 v16bf;
typedef __attribute__((ext_vector_type(8)))  float  v8f;

#define NN   50000
#define EE   800000
#define HIDD 64
#define LLL  4
#define KKK  2
#define BBB  64
#define ETOT (EE + NN)            // 850000 edges incl. self loops
#define WPB  8                    // waves per 256-thread block

// ---- order-preserving float<->uint encoding for atomicMax on floats ----
static __device__ __forceinline__ unsigned enc_f(float f) {
    unsigned u = __float_as_uint(f);
    return (u & 0x80000000u) ? ~u : (u | 0x80000000u);
}
static __device__ __forceinline__ float dec_f(unsigned u) {
    return __uint_as_float((u & 0x80000000u) ? (u & 0x7FFFFFFFu) : ~u);
}
#define ENC_NEG_INF 0x007FFFFFu   // enc_f(-inf)

// ---------------- init / convert kernels ----------------
__global__ void k_convert_x(const float* __restrict__ x, float* __restrict__ h,
                            __hip_bfloat16* __restrict__ hb, int n) {
    int i = blockIdx.x * blockDim.x + threadIdx.x;
    if (i < n) { float v = x[i]; h[i] = v; hb[i] = __float2bfloat16(v); }
}

__global__ void k_convert_w(const float* __restrict__ Wn,
                            __hip_bfloat16* __restrict__ Wb, int n) {
    int i = blockIdx.x * blockDim.x + threadIdx.x;
    if (i < n) Wb[i] = __float2bfloat16(Wn[i]);
}

__global__ void k_layer_init(float* __restrict__ acc, unsigned* __restrict__ mx,
                             float* __restrict__ dn) {
    int i = blockIdx.x * blockDim.x + threadIdx.x;
    if (i < NN * HIDD) acc[i] = 0.0f;
    if (i < KKK * NN) { mx[i] = ENC_NEG_INF; dn[i] = 0.0f; }
}

__global__ void k_g_init(float* __restrict__ g) {
    int i = blockIdx.x * blockDim.x + threadIdx.x;
    if (i < BBB * HIDD) g[i] = 0.0f;
}

// ---------------- WMMA GEMM: xp[k] = h @ Wn[l,k]  (bf16 -> f32) ----------------
// grid.x = N/16 = 3125 M-tiles, block = 64 threads = 2 waves (wave == head)
__global__ __launch_bounds__(64)
void k_gemm(const __hip_bfloat16* __restrict__ hb,     // [N][64] bf16
            const __hip_bfloat16* __restrict__ Wb_l,   // [K][64][64] bf16
            float* __restrict__ xp)                    // [K][N][64] f32
{
    const int lane = threadIdx.x & 31;
    const int head = threadIdx.x >> 5;
    const int m0   = blockIdx.x * 16;

    const __hip_bfloat16* W = Wb_l + (size_t)head * HIDD * HIDD;
    float* out = xp + (size_t)head * NN * HIDD;

    const int mrow = m0 + (lane & 15);
    const int kb8  = (lane >> 4);          // 0: K-base 0, 1: K-base 8 (in halves/8)

    // A fragments for the two 16x32 k-chunks (ISA 16-bit A layout)
    union { uint4 q[2]; v16bf v; } A[2];
    const uint4* hrow = reinterpret_cast<const uint4*>(hb + (size_t)mrow * HIDD);
    A[0].q[0] = hrow[0 + kb8];   // K = base .. base+7        (kc = 0)
    A[0].q[1] = hrow[2 + kb8];   // K = 16+base .. 16+base+7
    A[1].q[0] = hrow[4 + kb8];   // K = 32+base ..            (kc = 32)
    A[1].q[1] = hrow[6 + kb8];

    v8f c[4];
    #pragma unroll
    for (int t = 0; t < 4; ++t) c[t] = (v8f){0.f,0.f,0.f,0.f,0.f,0.f,0.f,0.f};

    #pragma unroll
    for (int kc = 0; kc < 2; ++kc) {
        // B-fragment: lane holds column N = lane%16, 16 K values in table order
        const unsigned short* wr =
            reinterpret_cast<const unsigned short*>(W + (size_t)(kc * 32) * HIDD + (lane & 15));
        #pragma unroll
        for (int t = 0; t < 4; ++t) {
            union { unsigned short e[16]; v16bf v; } Bf;
            #pragma unroll
            for (int s = 0; s < 16; ++s)
                Bf.e[s] = wr[(size_t)((s < 8 ? 8 * (lane >> 4) + s : 16 + 8 * (lane >> 4) + (s - 8)) * HIDD + t * 16)];
            c[t] = __builtin_amdgcn_wmma_f32_16x16x32_bf16(
                false, A[kc].v, false, Bf.v, (short)0, c[t], false, false);
        }
    }

    const int nlane = lane & 15;
    const int mhi   = (lane >> 4) * 8;
    #pragma unroll
    for (int t = 0; t < 4; ++t)
        #pragma unroll
        for (int r = 0; r < 8; ++r)
            out[(size_t)(m0 + mhi + r) * HIDD + t * 16 + nlane] = c[t][r];
}

// ---------------- edge phase ----------------
// wave per (edge, head): logit = sum_c a[c]*lrelu(xi+xj); atomic segment max
__global__ __launch_bounds__(256)
void k_logits(const float* __restrict__ xp, const int* __restrict__ ei,
              const float* __restrict__ att_l, float* __restrict__ lg,
              unsigned* __restrict__ mx)
{
    const int lane = threadIdx.x & 31;
    const int wid  = blockIdx.x * WPB + (threadIdx.x >> 5);
    if (wid >= KKK * ETOT) return;
    const int head = wid / ETOT;
    const int e    = wid - head * ETOT;
    int row, col;
    if (e < EE) { row = ei[e]; col = ei[EE + e]; } else { row = col = e - EE; }

    const float* xph = xp + (size_t)head * NN * HIDD;
    const float* a   = att_l + head * HIDD;

    float s = 0.0f;
    #pragma unroll
    for (int t = 0; t < 2; ++t) {
        const int c = lane + 32 * t;
        float v = xph[(size_t)row * HIDD + c] + xph[(size_t)col * HIDD + c];
        v = (v > 0.0f) ? v : 0.2f * v;
        s += v * a[c];
    }
    #pragma unroll
    for (int off = 16; off > 0; off >>= 1) s += __shfl_xor(s, off, 32);
    if (lane == 0) {
        lg[wid] = s;
        atomicMax(&mx[head * NN + col], enc_f(s));
    }
}

// thread per (edge, head): ex = exp(logit - max); segment-sum denominator
__global__ __launch_bounds__(256)
void k_expsum(const int* __restrict__ ei, float* __restrict__ lg,
              const unsigned* __restrict__ mx, float* __restrict__ dn)
{
    int idx = blockIdx.x * blockDim.x + threadIdx.x;
    if (idx >= KKK * ETOT) return;
    const int head = idx / ETOT;
    const int e    = idx - head * ETOT;
    const int col  = (e < EE) ? ei[EE + e] : e - EE;
    const float m  = dec_f(mx[head * NN + col]);
    const float ex = __expf(lg[idx] - m);
    lg[idx] = ex;
    atomicAdd(&dn[head * NN + col], ex);
}

// wave per (edge, head): acc[col] += alpha * xp[head][col]  (both heads summed)
__global__ __launch_bounds__(256)
void k_scatter(const int* __restrict__ ei, const float* __restrict__ lg,
               const float* __restrict__ dn, const float* __restrict__ xp,
               float* __restrict__ acc)
{
    const int lane = threadIdx.x & 31;
    const int wid  = blockIdx.x * WPB + (threadIdx.x >> 5);
    if (wid >= KKK * ETOT) return;
    const int head = wid / ETOT;
    const int e    = wid - head * ETOT;
    const int col  = (e < EE) ? ei[EE + e] : e - EE;

    const float alpha = lg[wid] / dn[head * NN + col];
    const float* xj   = xp + (size_t)head * NN * HIDD + (size_t)col * HIDD;
    #pragma unroll
    for (int t = 0; t < 2; ++t) {
        const int c = lane + 32 * t;
        atomicAdd(&acc[(size_t)col * HIDD + c], alpha * xj[c]);
    }
}

// wave per node: residual + mean-over-heads + LayerNorm; emit f32 + bf16
__global__ __launch_bounds__(256)
void k_ln(float* __restrict__ h, const float* __restrict__ acc,
          const float* __restrict__ sc, const float* __restrict__ bi,
          __hip_bfloat16* __restrict__ hb)
{
    const int lane = threadIdx.x & 31;
    const int n    = blockIdx.x * WPB + (threadIdx.x >> 5);
    if (n >= NN) return;
    const size_t base = (size_t)n * HIDD;

    float v0 = h[base + lane]      + 0.5f * acc[base + lane];       // 1/K = 0.5
    float v1 = h[base + lane + 32] + 0.5f * acc[base + lane + 32];

    float s = v0 + v1;
    #pragma unroll
    for (int off = 16; off > 0; off >>= 1) s += __shfl_xor(s, off, 32);
    const float mu = s * (1.0f / HIDD);

    const float d0 = v0 - mu, d1 = v1 - mu;
    float q = d0 * d0 + d1 * d1;
    #pragma unroll
    for (int off = 16; off > 0; off >>= 1) q += __shfl_xor(q, off, 32);
    const float r = rsqrtf(q * (1.0f / HIDD) + 1e-5f);

    const float o0 = d0 * r * sc[lane]      + bi[lane];
    const float o1 = d1 * r * sc[lane + 32] + bi[lane + 32];
    h[base + lane]       = o0;  hb[base + lane]      = __float2bfloat16(o0);
    h[base + lane + 32]  = o1;  hb[base + lane + 32] = __float2bfloat16(o1);
}

// wave per node: global_add_pool into g[B][64]
__global__ __launch_bounds__(256)
void k_pool(const float* __restrict__ h, const int* __restrict__ batch,
            float* __restrict__ g)
{
    const int lane = threadIdx.x & 31;
    const int n    = blockIdx.x * WPB + (threadIdx.x >> 5);
    if (n >= NN) return;
    const int b = batch[n];
    atomicAdd(&g[b * HIDD + lane],      h[(size_t)n * HIDD + lane]);
    atomicAdd(&g[b * HIDD + lane + 32], h[(size_t)n * HIDD + lane + 32]);
}

// block per graph (32 threads): hid = relu(g@W1+b1); out = hid@W2 + b2
__global__ __launch_bounds__(32)
void k_mlp(const float* __restrict__ g, const float* __restrict__ W1,
           const float* __restrict__ b1, const float* __restrict__ W2,
           const float* __restrict__ b2, float* __restrict__ out)
{
    const int b = blockIdx.x;
    const int j = threadIdx.x;        // 0..31
    float s = b1[j];
    #pragma unroll
    for (int i = 0; i < HIDD; ++i) s += g[b * HIDD + i] * W1[i * 32 + j];
    s = (s > 0.0f) ? s : 0.0f;
    float p = s * W2[j];
    #pragma unroll
    for (int off = 16; off > 0; off >>= 1) p += __shfl_xor(p, off, 32);
    if (j == 0) out[b] = p + b2[0];
}

// ---------------- host launcher ----------------
extern "C" void kernel_launch(void* const* d_in, const int* in_sizes, int n_in,
                              void* d_out, int out_size, void* d_ws, size_t ws_size,
                              hipStream_t stream) {
    const float* x     = (const float*)d_in[0];   // [N,64]
    const int*   ei    = (const int*)  d_in[1];   // [2,E]
    const int*   batch = (const int*)  d_in[2];   // [N]
    const float* Wn    = (const float*)d_in[3];   // [L,K,64,64]
    const float* att   = (const float*)d_in[4];   // [L,K,64]
    const float* lns   = (const float*)d_in[5];   // [L,64]
    const float* lnb   = (const float*)d_in[6];   // [L,64]
    const float* W1    = (const float*)d_in[7];   // [64,32]
    const float* b1    = (const float*)d_in[8];   // [32]
    const float* W2    = (const float*)d_in[9];   // [32,1]
    const float* b2    = (const float*)d_in[10];  // [1]
    float* out = (float*)d_out;

    // carve workspace (256B aligned chunks)
    char* p = (char*)d_ws;
    auto carve = [&](size_t bytes) -> void* {
        void* r = (void*)p;
        p += (bytes + 255) & ~(size_t)255;
        return r;
    };
    float*          h   = (float*)         carve((size_t)NN * HIDD * 4);
    __hip_bfloat16* hb  = (__hip_bfloat16*)carve((size_t)NN * HIDD * 2);
    __hip_bfloat16* Wb  = (__hip_bfloat16*)carve((size_t)LLL * KKK * HIDD * HIDD * 2);
    float*          xp  = (float*)         carve((size_t)KKK * NN * HIDD * 4);
    float*          lg  = (float*)         carve((size_t)KKK * ETOT * 4);
    unsigned*       mx  = (unsigned*)      carve((size_t)KKK * NN * 4);
    float*          dn  = (float*)         carve((size_t)KKK * NN * 4);
    float*          acc = (float*)         carve((size_t)NN * HIDD * 4);
    float*          g   = (float*)         carve((size_t)BBB * HIDD * 4);

    const int nh        = NN * HIDD;                       // 3.2M
    const int edgeWaves = KKK * ETOT;                      // 1.7M
    const int edgeBlk   = (edgeWaves + WPB - 1) / WPB;
    const int nodeBlk   = (NN + WPB - 1) / WPB;

    k_convert_x<<<(nh + 255) / 256, 256, 0, stream>>>(x, h, hb, nh);
    k_convert_w<<<(LLL * KKK * HIDD * HIDD + 255) / 256, 256, 0, stream>>>(
        Wn, Wb, LLL * KKK * HIDD * HIDD);

    for (int l = 0; l < LLL; ++l) {
        const __hip_bfloat16* Wb_l  = Wb + (size_t)l * KKK * HIDD * HIDD;
        const float*          att_l = att + (size_t)l * KKK * HIDD;
        const float*          sc_l  = lns + (size_t)l * HIDD;
        const float*          bi_l  = lnb + (size_t)l * HIDD;

        k_layer_init<<<(nh + 255) / 256, 256, 0, stream>>>(acc, mx, dn);
        k_gemm<<<NN / 16, 64, 0, stream>>>(hb, Wb_l, xp);
        k_logits<<<edgeBlk, 256, 0, stream>>>(xp, ei, att_l, lg, mx);
        k_expsum<<<(edgeWaves + 255) / 256, 256, 0, stream>>>(ei, lg, mx, dn);
        k_scatter<<<edgeBlk, 256, 0, stream>>>(ei, lg, dn, xp, acc);
        k_ln<<<nodeBlk, 256, 0, stream>>>(h, acc, sc_l, bi_l, hb);
    }

    k_g_init<<<(BBB * HIDD + 255) / 256, 256, 0, stream>>>(g);
    k_pool<<<nodeBlk, 256, 0, stream>>>(h, batch, g);
    k_mlp<<<BBB, 32, 0, stream>>>(g, W1, b1, W2, b2, out);
}
